// RW_Module_6854767805027
// MI455X (gfx1250) — matrix operand order repaired and checked by reading, not compiled, not run on hardware
//
#include <hip/hip_runtime.h>
#include <hip/hip_bf16.h>

// ---------------- problem constants ----------------
#define BATCH   4
#define C_IN    512
#define CQK     64
#define OTOT    640      // 64 q + 64 k + 512 v
#define HIMG    129
#define HW      65       // downsampled h=w
#define NPIX    4225     // 65*65
#define NPAD    4288     // multiple of 16/32/64 >= NPIX
#define HWIMG   (HIMG*HIMG)

typedef __attribute__((ext_vector_type(16))) _Float16 v16h;
typedef __attribute__((ext_vector_type(8)))  _Float16 v8h;
typedef __attribute__((ext_vector_type(8)))  float    v8f;

// ---------------- WMMA helpers (gfx1250, wave32) ----------------
// A (16x32 f16, row-major source): lanes 0-15 = rows M0..15 hold K0..15;
// lanes 16-31 = rows M0..15 hold K16..31.  rowp pre-offset by koff (=0/16).
__device__ __forceinline__ v16h load_a_row(const _Float16* __restrict__ rowp, int k0) {
    v8h lo = *reinterpret_cast<const v8h*>(rowp + k0);
    v8h hi = *reinterpret_cast<const v8h*>(rowp + k0 + 8);
    v16h a;
#pragma unroll
    for (int i = 0; i < 8; ++i) { a[i] = lo[i]; a[i + 8] = hi[i]; }
    return a;
}

// B (32x16 f16, column-major source): lanes 0-15 = cols N0..15 hold K0..15;
// lanes 16-31 = cols N0..15 hold K16..31.  colp pre-offset by kb (=0/16).
__device__ __forceinline__ v16h load_b_col(const _Float16* __restrict__ colp, int k0) {
    return *reinterpret_cast<const v16h*>(colp + k0);
}

__device__ __forceinline__ v8f wmma_f16(v16h a, v16h b, v8f c) {
    return __builtin_amdgcn_wmma_f32_16x16x32_f16(false, a, false, b, (short)0, c, false, false);
}

// ---------------- kernel 1: pack weights to f16 [640][512] ----------------
__global__ void convert_weights(const float* __restrict__ Wq, const float* __restrict__ Wk,
                                const float* __restrict__ Wv, _Float16* __restrict__ Wc) {
    int idx = blockIdx.x * blockDim.x + threadIdx.x;       // 640*512 threads
    int o = idx / C_IN, ci = idx % C_IN;
    float w;
    if (o < CQK)            w = Wq[o * C_IN + ci];
    else if (o < 2 * CQK)   w = Wk[(o - CQK) * C_IN + ci];
    else                    w = Wv[(o - 2 * CQK) * C_IN + ci];
    Wc[idx] = (_Float16)w;
}

// ---------------- kernel 2: downsample (exact ::2 subsample) + transpose ----------------
// xsT[b][n][c] f16, rows n >= NPIX zero-filled.
__global__ void downsample_pack(const float* __restrict__ x, _Float16* __restrict__ xsT) {
    __shared__ float tile[32][33];
    int b  = blockIdx.z;
    int c0 = blockIdx.y * 32;
    int n0 = blockIdx.x * 32;
    int tx = threadIdx.x & 31;
    int ty = threadIdx.x >> 5;                              // 0..7
    const float* xb = x + (size_t)b * C_IN * HWIMG;
#pragma unroll
    for (int rr = 0; rr < 4; ++rr) {
        int c = c0 + ty + rr * 8;
        int n = n0 + tx;
        float v = 0.f;
        if (n < NPIX) {
            int iy = n / HW, ix = n % HW;
            v = xb[(size_t)c * HWIMG + (2 * iy) * HIMG + (2 * ix)];
        }
        tile[ty + rr * 8][tx] = v;
    }
    __syncthreads();
    _Float16* ob = xsT + (size_t)b * NPAD * C_IN;
#pragma unroll
    for (int rr = 0; rr < 4; ++rr) {
        int n = n0 + ty + rr * 8;
        int c = c0 + tx;
        ob[(size_t)n * C_IN + c] = (_Float16)tile[tx][ty + rr * 8];
    }
}

// ---------------- kernel 3: fused q/k/v 1x1-conv GEMM (32x64 wave tile) ----------------
// D[o][n] = sum_c Wc[o][c]*xsT[n][c] + bias. q,k stored pixel-major; v channel-major.
__global__ void qkv_gemm(const _Float16* __restrict__ Wc,
                         const float* __restrict__ bq, const float* __restrict__ bk,
                         const float* __restrict__ bv,
                         const _Float16* __restrict__ xsT,
                         _Float16* __restrict__ qT, _Float16* __restrict__ kT,
                         _Float16* __restrict__ vM) {
    int wid  = (blockIdx.x * blockDim.x + threadIdx.x) >> 5;
    int lane = threadIdx.x & 31;
    const int tO = OTOT / 32, tN = NPAD / 64;               // 20, 67
    int per_b = tO * tN;
    int b = wid / per_b;
    if (b >= BATCH) return;
    int r  = wid % per_b;
    int o0 = (r / tN) * 32;
    int n0 = (r % tN) * 64;

    int koff = (lane >> 4) * 16;
    int kb   = (lane >> 4) * 16;
    const _Float16* ar0 = Wc + (size_t)(o0 + (lane & 15)) * C_IN + koff;
    const _Float16* ar1 = ar0 + (size_t)16 * C_IN;
    const _Float16* bb  = xsT + ((size_t)b * NPAD + n0 + (lane & 15)) * C_IN + kb;

    v8f acc[2][4];
#pragma unroll
    for (int i = 0; i < 2; ++i)
#pragma unroll
        for (int j = 0; j < 4; ++j) acc[i][j] = (v8f){};

#pragma unroll 2
    for (int k0 = 0; k0 < C_IN; k0 += 32) {
        v16h a0 = load_a_row(ar0, k0);
        v16h a1 = load_a_row(ar1, k0);
#pragma unroll
        for (int j = 0; j < 4; ++j) {
            v16h bm = load_b_col(bb + (size_t)(j * 16) * C_IN, k0);
            acc[0][j] = wmma_f16(a0, bm, acc[0][j]);
            acc[1][j] = wmma_f16(a1, bm, acc[1][j]);
        }
    }

    int rbase = (lane >> 4) * 8;
#pragma unroll
    for (int i = 0; i < 2; ++i) {
#pragma unroll
        for (int j = 0; j < 4; ++j) {
            int col = n0 + j * 16 + (lane & 15);
#pragma unroll
            for (int rr = 0; rr < 8; ++rr) {
                int o = o0 + i * 16 + rbase + rr;
                float bias = (o < CQK) ? bq[o] : (o < 2 * CQK ? bk[o - CQK] : bv[o - 2 * CQK]);
                float val  = acc[i][j][rr] + bias;
                if (o < CQK)
                    qT[((size_t)b * NPAD + col) * CQK + o] = (_Float16)val;
                else if (o < 2 * CQK)
                    kT[((size_t)b * NPAD + col) * CQK + (o - CQK)] = (_Float16)val;
                else
                    vM[(size_t)b * C_IN * NPAD + (size_t)(o - 2 * CQK) * NPAD + col] = (_Float16)val;
            }
        }
    }
}

// ---------------- kernel 4: energy = Q^T K  (fp32 out to d_out), 32x64 wave tile ----------------
__global__ void energy_gemm(const _Float16* __restrict__ qT, const _Float16* __restrict__ kT,
                            float* __restrict__ energy) {
    int wid  = (blockIdx.x * blockDim.x + threadIdx.x) >> 5;
    int lane = threadIdx.x & 31;
    const int tn = NPAD / 32, tm = NPAD / 64;               // 134, 67
    int per_b = tn * tm;
    int b = wid / per_b;
    if (b >= BATCH) return;
    int r  = wid % per_b;
    int n0 = (r / tm) * 32;
    int m0 = (r % tm) * 64;

    int koff = (lane >> 4) * 16;
    int kb   = (lane >> 4) * 16;
    const _Float16* ar0 = qT + ((size_t)b * NPAD + n0 + (lane & 15)) * CQK + koff;
    const _Float16* ar1 = ar0 + (size_t)16 * CQK;
    const _Float16* bb  = kT + ((size_t)b * NPAD + m0 + (lane & 15)) * CQK + kb;

    v8f acc[2][4];
#pragma unroll
    for (int i = 0; i < 2; ++i)
#pragma unroll
        for (int j = 0; j < 4; ++j) acc[i][j] = (v8f){};

#pragma unroll
    for (int k0 = 0; k0 < CQK; k0 += 32) {
        v16h a0 = load_a_row(ar0, k0);
        v16h a1 = load_a_row(ar1, k0);
#pragma unroll
        for (int j = 0; j < 4; ++j) {
            v16h bm = load_b_col(bb + (size_t)(j * 16) * CQK, k0);
            acc[0][j] = wmma_f16(a0, bm, acc[0][j]);
            acc[1][j] = wmma_f16(a1, bm, acc[1][j]);
        }
    }

    float* E = energy + (size_t)b * NPIX * NPIX;
    int rbase = (lane >> 4) * 8;
#pragma unroll
    for (int i = 0; i < 2; ++i) {
#pragma unroll
        for (int j = 0; j < 4; ++j) {
            int col = m0 + j * 16 + (lane & 15);
            if (col >= NPIX) continue;
#pragma unroll
            for (int rr = 0; rr < 8; ++rr) {
                int row = n0 + i * 16 + rbase + rr;
                if (row < NPIX) E[(size_t)row * NPIX + col] = acc[i][j][rr];
            }
        }
    }
}

// ---------------- kernel 5: row softmax (register-cached) -> f16 att, padded ----------------
__global__ void softmax_rows(const float* __restrict__ energy, _Float16* __restrict__ att) {
    const int T = 256;
    const int PER = (NPIX + T - 1) / T;                     // 17
    int rowid = blockIdx.x;
    int b = rowid / NPAD, n = rowid % NPAD;
    _Float16* arow = att + ((size_t)b * NPAD + n) * NPAD;
    int t = threadIdx.x;
    if (n >= NPIX) {
        for (int m = t; m < NPAD; m += T) arow[m] = (_Float16)0.f;
        return;
    }
    const float* erow = energy + ((size_t)b * NPIX + n) * NPIX;
    float ev[PER];
#pragma unroll
    for (int i = 0; i < PER; ++i) {
        int m = t + i * T;
        ev[i] = (m < NPIX) ? erow[m] : -1e30f;
    }
    __shared__ float red[8];
    int lane = t & 31, w = t >> 5;
    float mx = -1e30f;
#pragma unroll
    for (int i = 0; i < PER; ++i) mx = fmaxf(mx, ev[i]);
#pragma unroll
    for (int off = 16; off > 0; off >>= 1) mx = fmaxf(mx, __shfl_xor(mx, off, 32));
    if (lane == 0) red[w] = mx;
    __syncthreads();
    float bmx = red[0];
#pragma unroll
    for (int i = 1; i < 8; ++i) bmx = fmaxf(bmx, red[i]);
    __syncthreads();
    float s = 0.f;
#pragma unroll
    for (int i = 0; i < PER; ++i) { ev[i] = __expf(ev[i] - bmx); s += ev[i]; }
#pragma unroll
    for (int off = 16; off > 0; off >>= 1) s += __shfl_xor(s, off, 32);
    if (lane == 0) red[w] = s;
    __syncthreads();
    float tot = 0.f;
#pragma unroll
    for (int i = 0; i < 8; ++i) tot += red[i];
    float inv = 1.0f / tot;
#pragma unroll
    for (int i = 0; i < PER; ++i) {
        int m = t + i * T;
        if (m < NPIX) arow[m] = (_Float16)(ev[i] * inv);
    }
    for (int m = NPIX + t; m < NPAD; m += T) arow[m] = (_Float16)0.f;
}

// ---------------- kernel 6: out2 = V * Att^T (dominant GEMM), 32x64 wave tile ----------------
__global__ void out_gemm(const _Float16* __restrict__ vM, const _Float16* __restrict__ att,
                         float* __restrict__ out2) {
    int wid  = (blockIdx.x * blockDim.x + threadIdx.x) >> 5;
    int lane = threadIdx.x & 31;
    const int tC = C_IN / 32, tN = NPAD / 64;               // 16, 67
    int per_b = tC * tN;
    int b = wid / per_b;
    if (b >= BATCH) return;
    int r  = wid % per_b;
    int c0 = (r / tN) * 32;
    int n0 = (r % tN) * 64;

    int koff = (lane >> 4) * 16;
    int kb   = (lane >> 4) * 16;
    const _Float16* ar0 = vM + (size_t)b * C_IN * NPAD + (size_t)(c0 + (lane & 15)) * NPAD + koff;
    const _Float16* ar1 = ar0 + (size_t)16 * NPAD;
    const _Float16* bb  = att + (size_t)b * NPAD * NPAD + (size_t)(n0 + (lane & 15)) * NPAD + kb;

    v8f acc[2][4];
#pragma unroll
    for (int i = 0; i < 2; ++i)
#pragma unroll
        for (int j = 0; j < 4; ++j) acc[i][j] = (v8f){};

    for (int k0 = 0; k0 < NPAD; k0 += 32) {
        __builtin_prefetch(ar0 + k0 + 512, 0, 1);
        __builtin_prefetch(ar1 + k0 + 512, 0, 1);
        v16h a0 = load_a_row(ar0, k0);
        v16h a1 = load_a_row(ar1, k0);
#pragma unroll
        for (int j = 0; j < 4; ++j) {
            v16h bm = load_b_col(bb + (size_t)(j * 16) * NPAD, k0);
            acc[0][j] = wmma_f16(a0, bm, acc[0][j]);
            acc[1][j] = wmma_f16(a1, bm, acc[1][j]);
        }
    }

    int rbase = (lane >> 4) * 8;
#pragma unroll
    for (int i = 0; i < 2; ++i) {
#pragma unroll
        for (int j = 0; j < 4; ++j) {
            int col = n0 + j * 16 + (lane & 15);
            if (col >= NPIX) continue;
#pragma unroll
            for (int rr = 0; rr < 8; ++rr)
                out2[((size_t)b * C_IN + c0 + i * 16 + rbase + rr) * NPIX + col] = acc[i][j][rr];
        }
    }
}

// ---------------- kernel 7: exact 2x upsample + gamma*up + x ----------------
__global__ void upsample_residual(const float* __restrict__ out2, const float* __restrict__ x,
                                  const float* __restrict__ gamma, float* __restrict__ outp) {
    size_t i = (size_t)blockIdx.x * blockDim.x + threadIdx.x;
    const size_t total = (size_t)BATCH * C_IN * HWIMG;
    if (i >= total) return;
    int X = (int)(i % HIMG);
    int Y = (int)((i / HIMG) % HIMG);
    size_t bc = i / HWIMG;                                  // b*C + c
    int iy0 = Y >> 1, ix0 = X >> 1;
    const float* g = out2 + bc * NPIX;
    float v;
    if (((Y & 1) == 0) && ((X & 1) == 0))
        v = g[iy0 * HW + ix0];
    else if ((Y & 1) == 0)
        v = 0.5f * (g[iy0 * HW + ix0] + g[iy0 * HW + ix0 + 1]);
    else if ((X & 1) == 0)
        v = 0.5f * (g[iy0 * HW + ix0] + g[(iy0 + 1) * HW + ix0]);
    else
        v = 0.25f * (g[iy0 * HW + ix0] + g[iy0 * HW + ix0 + 1] +
                     g[(iy0 + 1) * HW + ix0] + g[(iy0 + 1) * HW + ix0 + 1]);
    outp[i] = gamma[0] * v + x[i];
}

// ---------------- launch ----------------
extern "C" void kernel_launch(void* const* d_in, const int* in_sizes, int n_in,
                              void* d_out, int out_size, void* d_ws, size_t ws_size,
                              hipStream_t stream) {
    const float* x  = (const float*)d_in[0];
    const float* Wq = (const float*)d_in[1];
    const float* bq = (const float*)d_in[2];
    const float* Wk = (const float*)d_in[3];
    const float* bk = (const float*)d_in[4];
    const float* Wv = (const float*)d_in[5];
    const float* bv = (const float*)d_in[6];
    const float* gm = (const float*)d_in[7];

    float* outp   = (float*)d_out;
    float* energy = outp + (size_t)BATCH * C_IN * HWIMG;    // second tuple element

    // workspace layout (all 256B-aligned)
    char* ws = (char*)d_ws;
    constexpr size_t SZ_WC   = (size_t)OTOT * C_IN * 2;                 //   0.66 MB
    constexpr size_t SZ_XST  = (size_t)BATCH * NPAD * C_IN * 2;         //  17.6 MB
    constexpr size_t SZ_QT   = (size_t)BATCH * NPAD * CQK * 2;          //   2.2 MB
    constexpr size_t SZ_VM   = (size_t)BATCH * C_IN * NPAD * 2;         //  17.6 MB
    constexpr size_t SZ_ATT  = (size_t)BATCH * NPAD * NPAD * 2;         // 147.1 MB
    size_t off = 0;
    _Float16* Wc  = (_Float16*)(ws + off); off += SZ_WC;
    _Float16* xsT = (_Float16*)(ws + off); off += SZ_XST;
    _Float16* qT  = (_Float16*)(ws + off); off += SZ_QT;
    _Float16* kT  = (_Float16*)(ws + off); off += SZ_QT;
    _Float16* vM  = (_Float16*)(ws + off); off += SZ_VM;
    _Float16* attp= (_Float16*)(ws + off); off += SZ_ATT;
    float*    out2= (float*)   (ws + off);                              //  34.6 MB

    convert_weights<<<(OTOT * C_IN) / 256, 256, 0, stream>>>(Wq, Wk, Wv, Wc);
    downsample_pack<<<dim3(NPAD / 32, C_IN / 32, BATCH), 256, 0, stream>>>(x, xsT);
    qkv_gemm<<<(BATCH * (OTOT / 32) * (NPAD / 64)) / 8, 256, 0, stream>>>(
        Wc, bq, bk, bv, xsT, qT, kT, vM);
    energy_gemm<<<(BATCH * (NPAD / 32) * (NPAD / 64)) / 8, 256, 0, stream>>>(qT, kT, energy);
    softmax_rows<<<BATCH * NPAD, 256, 0, stream>>>(energy, attp);
    out_gemm<<<(BATCH * (C_IN / 32) * (NPAD / 64)) / 8, 256, 0, stream>>>(vM, attp, out2);
    {
        size_t total = (size_t)BATCH * C_IN * HWIMG;
        upsample_residual<<<(unsigned)((total + 255) / 256), 256, 0, stream>>>(out2, x, gm, outp);
    }
}
